// ForwardModelLSTM_SD_22136261444393
// MI455X (gfx1250) — matrix operand order repaired and checked by reading, not compiled, not run on hardware
//
#include <hip/hip_runtime.h>
#include <math.h>

// ---------------------------------------------------------------------------
// LSTM forward, MI455X (gfx1250, wave32).
//   T=4096 steps, IN=1088, H=2048, 4H=8192, S=1024.
// Phases:
//   0) fp32 -> bf16 convert kernels (packed, bandwidth-bound)
//   1) x   = inputs @ W_embed^T + b_embed        (WMMA bf16 GEMM, f32 accum)
//   2) gx  = x @ W_ih^T + b_ih                   (WMMA bf16 GEMM, f32 accum)
//   3) sequential LSTM scan over t (persistent kernel, grid barrier / step,
//      wave-cooperative coalesced dots + shuffle reduction)
//   4) logits = hs @ W_out^T + b_out             (WMMA bf16 GEMM, f32 accum)
//   5) softmax rows -> d_out [T,1,S] fp32
// ---------------------------------------------------------------------------

#define LSTM_T  4096
#define LSTM_IN 1088
#define LSTM_H  2048
#define LSTM_4H 8192
#define LSTM_S  1024

typedef __attribute__((ext_vector_type(16))) __bf16 v16bf;
typedef __attribute__((ext_vector_type(8)))  float  v8f;

__device__ __forceinline__ unsigned bfround(unsigned u) {
    return u + 0x7FFFu + ((u >> 16) & 1u);          // round-to-nearest-even
}
__device__ __forceinline__ unsigned pack2bf(float lo, float hi) {
    unsigned ul = __float_as_uint(lo), uh = __float_as_uint(hi);
    return (bfround(ul) >> 16) | (bfround(uh) & 0xFFFF0000u);
}
__device__ __forceinline__ float sigmoidf_dev(float x) {
    return 1.0f / (1.0f + __expf(-x));
}

// ---------------------------------------------------------------------------
// fp32 -> bf16 convert, 8 elements / thread, uint4 (8 x bf16) stores.
// n8 = element_count / 8 (all counts here divisible by 2048).
// ---------------------------------------------------------------------------
__global__ __launch_bounds__(256)
void f32_to_bf16(const float* __restrict__ in, __bf16* __restrict__ out,
                 int n8) {
    int i = blockIdx.x * 256 + threadIdx.x;
    if (i >= n8) return;
    const float4* p = (const float4*)in + (size_t)i * 2;
    float4 a = p[0];
    float4 b = p[1];
    uint4 v;
    v.x = pack2bf(a.x, a.y);
    v.y = pack2bf(a.z, a.w);
    v.z = pack2bf(b.x, b.y);
    v.w = pack2bf(b.z, b.w);
    ((uint4*)out)[i] = v;
}

// ---------------------------------------------------------------------------
// GEMM: C[M,N] = X[M,K] @ W[N,K]^T + bias[N]    (X, W already bf16)
// Block: 256 threads (8 waves). Block tile: 64 (M) x 128 (N).
// Waves arranged 2 (M) x 4 (N); each wave holds 2 A-frags x 2 B-frags and
// issues 4 x v_wmma_f32_16x16x32_bf16 per 32-wide K step (full frag reuse).
// Staging: global_load_b128 -> ds_store_b128; next tiles prefetched
// (global_prefetch_b8) while WMMAs execute.
// M % 64 == 0, N % 128 == 0, K % 32 == 0 (true for all three GEMMs here).
// ---------------------------------------------------------------------------
__global__ __launch_bounds__(256)
void gemm_bias_wmma(const __bf16* __restrict__ X, const __bf16* __restrict__ W,
                    const float* __restrict__ bias, float* __restrict__ C,
                    int M, int N, int K) {
    __shared__ __bf16 As[64 * 32];    // [m 0..63][k 0..31]
    __shared__ __bf16 Bs[128 * 32];   // [n 0..127][k 0..31]  (row of W = n)

    const int tid  = threadIdx.x;
    const int lane = tid & 31;
    const int wave = tid >> 5;
    const int m0   = blockIdx.y * 64;
    const int n0   = blockIdx.x * 128;
    const int mb   = (wave >> 2) * 32;   // 0,32
    const int nb   = (wave & 3) * 32;    // 0,32,64,96
    const int half = lane >> 4;          // 0..1
    const int l16  = lane & 15;

    // Staging coordinates: uint4 = 8 bf16; each 32-wide row = 4 uint4.
    const int ar = tid >> 2, aq = (tid & 3) * 8;        // A: 1 uint4 / thread
    const __bf16* __restrict__ Ap = X + (size_t)(m0 + ar) * K + aq;
    const __bf16* __restrict__ Bp0 = W + (size_t)(n0 + ar) * K + aq;
    const __bf16* __restrict__ Bp1 = W + (size_t)(n0 + 64 + ar) * K + aq;

    v8f acc00 = {}, acc01 = {}, acc10 = {}, acc11 = {};

    for (int k0 = 0; k0 < K; k0 += 32) {
        // Stage A tile 64x32 (256 uint4 total, 1 per thread)
        ((uint4*)As)[tid] = *(const uint4*)(Ap + k0);
        // Stage B tile 128x32 (512 uint4 total, 2 per thread)
        ((uint4*)Bs)[tid]       = *(const uint4*)(Bp0 + k0);
        ((uint4*)Bs)[tid + 256] = *(const uint4*)(Bp1 + k0);

        // Prefetch next K-step tiles into cache while WMMAs run
        if (k0 + 32 < K) {
            __builtin_prefetch(Ap + k0 + 32, 0, 3);
            __builtin_prefetch(Bp0 + k0 + 32, 0, 3);
            __builtin_prefetch(Bp1 + k0 + 32, 0, 3);
        }
        __syncthreads();

        // Gather WMMA fragments (contiguous k = half*16 + e in both A and B, wave32)
        v16bf a0, a1, b0, b1;
        #pragma unroll
        for (int e = 0; e < 16; ++e) {
            int ka = half * 16 + e;
            a0[e] = As[(mb + l16) * 32 + ka];
            a1[e] = As[(mb + 16 + l16) * 32 + ka];
            int kb = half * 16 + e;
            b0[e] = Bs[(nb + l16) * 32 + kb];
            b1[e] = Bs[(nb + 16 + l16) * 32 + kb];
        }

        acc00 = __builtin_amdgcn_wmma_f32_16x16x32_bf16(
                    false, a0, false, b0, (short)0, acc00, false, false);
        acc01 = __builtin_amdgcn_wmma_f32_16x16x32_bf16(
                    false, a0, false, b1, (short)0, acc01, false, false);
        acc10 = __builtin_amdgcn_wmma_f32_16x16x32_bf16(
                    false, a1, false, b0, (short)0, acc10, false, false);
        acc11 = __builtin_amdgcn_wmma_f32_16x16x32_bf16(
                    false, a1, false, b1, (short)0, acc11, false, false);
        __syncthreads();
    }

    // C/D layout: VGPR v -> M = v + half*8, N = l16
    const int nc0 = n0 + nb + l16;
    const int nc1 = nc0 + 16;
    const float bias0 = bias[nc0];
    const float bias1 = bias[nc1];
    #pragma unroll
    for (int v = 0; v < 8; ++v) {
        int mr0 = m0 + mb + v + half * 8;
        int mr1 = mr0 + 16;
        C[(size_t)mr0 * N + nc0] = acc00[v] + bias0;
        C[(size_t)mr0 * N + nc1] = acc01[v] + bias1;
        C[(size_t)mr1 * N + nc0] = acc10[v] + bias0;
        C[(size_t)mr1 * N + nc1] = acc11[v] + bias1;
    }
}

// ---------------------------------------------------------------------------
// Sequential LSTM scan. NWG=32 persistent workgroups of 512 threads (16
// waves). WG wg owns hidden units [wg*64, wg*64+64) -> 256 gate rows.
// Each wave owns 16 rows; the 32 lanes do a lane-strided float4 dot over
// LDS-staged h (coalesced 512B per issue) and reduce via wave32 shuffles.
// h state ping-pongs between two global buffers; one atomic grid barrier per
// step orders writes of h(t) before reads at t+1. c state lives in LDS.
// W_hh (67 MB fp32) is L2-resident across all 4096 steps (192 MB L2).
// ---------------------------------------------------------------------------
#define SCAN_NWG 32
#define SCAN_HC  64

__global__ __launch_bounds__(512)
void lstm_scan(const float* __restrict__ gx,    // [T][4H] (incl. b_ih)
               const float* __restrict__ W_hh,  // [4H][H]
               const float* __restrict__ b_hh,  // [4H]
               float* __restrict__ hbuf,        // [2][H], zeroed
               float* __restrict__ hs,          // [T][H]
               unsigned* __restrict__ bar) {    // zeroed
    __shared__ float h_sh[LSTM_H];
    __shared__ float gate_sh[4 * SCAN_HC];      // [q*64 + u]
    __shared__ float c_sh[SCAN_HC];

    const int tid  = threadIdx.x;               // 0..511
    const int lane = tid & 31;
    const int wave = tid >> 5;                  // 0..15
    const int wg   = blockIdx.x;

    if (tid < SCAN_HC) c_sh[tid] = 0.0f;

    for (int t = 0; t < LSTM_T; ++t) {
        const float4* __restrict__ hin =
            (const float4*)(hbuf + (size_t)(t & 1) * LSTM_H);
        float* __restrict__ hout = hbuf + (size_t)((t + 1) & 1) * LSTM_H;

        // Stage h(t-1) into LDS: 512 threads x 1 float4 = 2048 floats
        ((float4*)h_sh)[tid] = hin[tid];
        __syncthreads();

        // 16 gate rows per wave, lane-strided coalesced dot + shuffle reduce
        const float4* hv = (const float4*)h_sh;
        for (int rr = 0; rr < 16; ++rr) {
            int r   = wave * 16 + rr;                    // 0..255
            int q   = r >> 6;                            // gate 0..3
            int u   = r & 63;                            // unit in slice
            int row = q * LSTM_H + wg * SCAN_HC + u;     // gate row in [0,4H)
            const float4* __restrict__ Wr =
                (const float4*)(W_hh + (size_t)row * LSTM_H);
            float acc = 0.0f;
            #pragma unroll 4
            for (int i = lane; i < LSTM_H / 4; i += 32) {
                float4 w = Wr[i];
                float4 h = hv[i];
                acc = fmaf(w.x, h.x, acc);
                acc = fmaf(w.y, h.y, acc);
                acc = fmaf(w.z, h.z, acc);
                acc = fmaf(w.w, h.w, acc);
            }
            #pragma unroll
            for (int off = 16; off > 0; off >>= 1)
                acc += __shfl_down(acc, off, 32);
            if (lane == 0)
                gate_sh[r] = acc + gx[(size_t)t * LSTM_4H + row] + b_hh[row];
        }
        __syncthreads();

        if (tid < SCAN_HC) {
            float ig = sigmoidf_dev(gate_sh[0 * SCAN_HC + tid]);
            float fg = sigmoidf_dev(gate_sh[1 * SCAN_HC + tid]);
            float gg = tanhf(gate_sh[2 * SCAN_HC + tid]);
            float og = sigmoidf_dev(gate_sh[3 * SCAN_HC + tid]);
            float c  = fg * c_sh[tid] + ig * gg;
            c_sh[tid] = c;
            float h = og * tanhf(c);
            int kk = wg * SCAN_HC + tid;
            hout[kk] = h;
            hs[(size_t)t * LSTM_H + kk] = h;
        }
        __threadfence();
        __syncthreads();

        // Grid barrier: all WGs finished step t before anyone reads h(t)
        if (tid == 0) {
            atomicAdd(bar, 1u);
            unsigned target = (unsigned)(SCAN_NWG * (t + 1));
            while (atomicAdd(bar, 0u) < target) { }
        }
        __syncthreads();
        __threadfence();
    }
}

// ---------------------------------------------------------------------------
// Row softmax: one 256-thread block per row of S=1024.
// ---------------------------------------------------------------------------
__global__ __launch_bounds__(256)
void softmax_rows(const float* __restrict__ logits, float* __restrict__ out) {
    __shared__ float red[256];
    const int row = blockIdx.x;
    const int tid = threadIdx.x;
    const float* x = logits + (size_t)row * LSTM_S;

    float m = -INFINITY;
    for (int i = tid; i < LSTM_S; i += 256) m = fmaxf(m, x[i]);
    red[tid] = m; __syncthreads();
    for (int s = 128; s > 0; s >>= 1) {
        if (tid < s) red[tid] = fmaxf(red[tid], red[tid + s]);
        __syncthreads();
    }
    m = red[0]; __syncthreads();

    float sum = 0.0f;
    for (int i = tid; i < LSTM_S; i += 256) sum += __expf(x[i] - m);
    red[tid] = sum; __syncthreads();
    for (int s = 128; s > 0; s >>= 1) {
        if (tid < s) red[tid] += red[tid + s];
        __syncthreads();
    }
    float inv = 1.0f / red[0];

    for (int i = tid; i < LSTM_S; i += 256)
        out[(size_t)row * LSTM_S + i] = __expf(x[i] - m) * inv;
}

// ---------------------------------------------------------------------------
extern "C" void kernel_launch(void* const* d_in, const int* in_sizes, int n_in,
                              void* d_out, int out_size, void* d_ws, size_t ws_size,
                              hipStream_t stream) {
    (void)in_sizes; (void)n_in; (void)out_size; (void)ws_size;

    const float* inputs  = (const float*)d_in[0];  // [T, IN]
    const float* W_embed = (const float*)d_in[1];  // [H, IN]
    const float* b_embed = (const float*)d_in[2];  // [H]
    const float* W_ih    = (const float*)d_in[3];  // [4H, H]
    const float* b_ih    = (const float*)d_in[4];  // [4H]
    const float* W_hh    = (const float*)d_in[5];  // [4H, H]
    const float* b_hh    = (const float*)d_in[6];  // [4H]
    const float* W_out   = (const float*)d_in[7];  // [S, H]
    const float* b_out   = (const float*)d_in[8];  // [S]
    float* probs = (float*)d_out;                  // [T, 1, S]

    // ---- Workspace layout ----
    // fp32 region
    float* x      = (float*)d_ws;                          // T*H
    float* gx     = x      + (size_t)LSTM_T * LSTM_H;      // T*4H
    float* hs     = gx     + (size_t)LSTM_T * LSTM_4H;     // T*H
    float* logits = hs     + (size_t)LSTM_T * LSTM_H;      // T*S
    float* hbuf   = logits + (size_t)LSTM_T * LSTM_S;      // 2*H
    unsigned* bar = (unsigned*)(hbuf + 2 * LSTM_H);        // 1 (+3 pad)
    // bf16 region (16B aligned: all preceding counts are multiples of 4)
    __bf16* inputs_bf = (__bf16*)(bar + 4);
    __bf16* Wemb_bf   = inputs_bf + (size_t)LSTM_T * LSTM_IN;   // H*IN
    __bf16* x_bf      = Wemb_bf   + (size_t)LSTM_H * LSTM_IN;   // T*H
    __bf16* Wih_bf    = x_bf      + (size_t)LSTM_T * LSTM_H;    // 4H*H
    __bf16* hs_bf     = Wih_bf    + (size_t)LSTM_4H * LSTM_H;   // T*H
    __bf16* Wout_bf   = hs_bf     + (size_t)LSTM_T * LSTM_H;    // S*H

    // Zero h ping-pong buffers + grid-barrier counter (graph-capture safe)
    hipMemsetAsync(hbuf, 0, 2 * LSTM_H * sizeof(float) + 4 * sizeof(unsigned),
                   stream);

    // Phase 0: convert static operands to bf16
    {
        int n8;
        n8 = LSTM_T * LSTM_IN / 8;
        f32_to_bf16<<<n8 / 256, 256, 0, stream>>>(inputs, inputs_bf, n8);
        n8 = LSTM_H * LSTM_IN / 8;
        f32_to_bf16<<<n8 / 256, 256, 0, stream>>>(W_embed, Wemb_bf, n8);
        n8 = LSTM_4H * LSTM_H / 8;
        f32_to_bf16<<<n8 / 256, 256, 0, stream>>>(W_ih, Wih_bf, n8);
        n8 = LSTM_S * LSTM_H / 8;
        f32_to_bf16<<<n8 / 256, 256, 0, stream>>>(W_out, Wout_bf, n8);
    }

    // Phase 1: x = inputs @ W_embed^T + b_embed   (M=T, N=H, K=IN)
    gemm_bias_wmma<<<dim3(LSTM_H / 128, LSTM_T / 64), 256, 0, stream>>>(
        inputs_bf, Wemb_bf, b_embed, x, LSTM_T, LSTM_H, LSTM_IN);

    // Phase 1b: convert x to bf16
    {
        int n8 = LSTM_T * LSTM_H / 8;
        f32_to_bf16<<<n8 / 256, 256, 0, stream>>>(x, x_bf, n8);
    }

    // Phase 2: gx = x @ W_ih^T + b_ih             (M=T, N=4H, K=H)
    gemm_bias_wmma<<<dim3(LSTM_4H / 128, LSTM_T / 64), 256, 0, stream>>>(
        x_bf, Wih_bf, b_ih, gx, LSTM_T, LSTM_4H, LSTM_H);

    // Phase 3: sequential scan -> hs
    lstm_scan<<<SCAN_NWG, 512, 0, stream>>>(gx, W_hh, b_hh, hbuf, hs, bar);

    // Phase 3b: convert hs to bf16
    {
        int n8 = LSTM_T * LSTM_H / 8;
        f32_to_bf16<<<n8 / 256, 256, 0, stream>>>(hs, hs_bf, n8);
    }

    // Phase 4: logits = hs @ W_out^T + b_out      (M=T, N=S, K=H)
    gemm_bias_wmma<<<dim3(LSTM_S / 128, LSTM_T / 64), 256, 0, stream>>>(
        hs_bf, Wout_bf, b_out, logits, LSTM_T, LSTM_S, LSTM_H);

    // Phase 5: softmax rows -> probs
    softmax_rows<<<LSTM_T, 256, 0, stream>>>(logits, probs);
}
